// ModifiedEResGAT_17093969838498
// MI455X (gfx1250) — matrix operand order repaired and checked by reading, not compiled, not run on hardware
//
#include <hip/hip_runtime.h>
#include <hip/hip_bf16.h>

// ---------------------------------------------------------------------------
// Modified EResGAT (2-layer GAT with edge features) for MI455X / gfx1250.
// - Edge GEMM never materializes ef[E,H,C]; its epilogue computes
//   e_att[e,h] = dot(ef[e,h,:], att_edge[h,:]) and atomically accumulates the
//   self-loop numerator s_att[dst,h] (linearity of the dot == dot of segsum).
// - GEMMs: v_wmma_f32_16x16x32_f16 (wave32 WMMA), f32 accumulate.
// - Fused edge GEMM prefetches its A tile via GLOBAL_LOAD_ASYNC_TO_LDS_B128
//   (ASYNCcnt) and reads fragments from LDS.
// ---------------------------------------------------------------------------

typedef __attribute__((ext_vector_type(16))) _Float16 v16h;
typedef __attribute__((ext_vector_type(8)))  float    v8f;
typedef __attribute__((ext_vector_type(4)))  float    f4;

#define NN   50000
#define EE   800000
#define FIN  128
#define EDIM 64
#define HC   256
#define NH   4
#define CC   64
#define NEG_SLOPE 0.2f

static __device__ __forceinline__ unsigned enc_ord(float f) {
  unsigned u = __float_as_uint(f);
  return (u & 0x80000000u) ? ~u : (u | 0x80000000u);
}
static __device__ __forceinline__ float dec_ord(unsigned k) {
  unsigned u = (k & 0x80000000u) ? (k ^ 0x80000000u) : ~k;
  return __uint_as_float(u);
}
static __device__ __forceinline__ float lrelu(float x) {
  return x > 0.f ? x : NEG_SLOPE * x;
}

// ---------------------------------------------------------------------------
// Dense GEMM  D[M,256] = A[M,K] @ W[K,256]   (f32 in, f16 WMMA, f32 out)
// Block = 128 threads = 4 waves; wave w owns a 16x64 slab (4 wmma accs).
// M is a multiple of 16 (50000, 800000) so no tail handling.
// ---------------------------------------------------------------------------
__global__ __launch_bounds__(128)
void gemm_node_kernel(const float* __restrict__ A, const float* __restrict__ W,
                      float* __restrict__ D, int K) {
  const int lane = threadIdx.x & 31;
  const int wave = threadIdx.x >> 5;
  const int hi   = lane >> 4;
  const int l16  = lane & 15;
  const int n0   = wave * 64;
  const long arow = (long)blockIdx.x * 16 + l16;

  v8f acc[4] = {};
  for (int kk = 0; kk < K; kk += 32) {
    // ---- A fragment: lane holds row (l16), K = kk + {hi*8..+7, 16+hi*8..+7}
    const float* ap = A + arow * K + kk + hi * 8;
    f4 a0 = *(const f4*)(ap);
    f4 a1 = *(const f4*)(ap + 4);
    f4 a2 = *(const f4*)(ap + 16);
    f4 a3 = *(const f4*)(ap + 20);
    v16h af;
#pragma unroll
    for (int j = 0; j < 4; ++j) {
      af[j]      = (_Float16)a0[j];
      af[4 + j]  = (_Float16)a1[j];
      af[8 + j]  = (_Float16)a2[j];
      af[12 + j] = (_Float16)a3[j];
    }
    // ---- B fragments: lane holds col n0+t*16+l16; K = kk + {hi*8..+7, 16+hi*8..+7}
    const float* wp = W + (long)(kk + hi * 8) * HC + l16;
#pragma unroll
    for (int t = 0; t < 4; ++t) {
      const float* bp = wp + n0 + t * 16;
      v16h bf;
#pragma unroll
      for (int j = 0; j < 4; ++j) {
        bf[j]      = (_Float16)bp[(long)j * HC];
        bf[4 + j]  = (_Float16)bp[(long)(4 + j) * HC];
        bf[8 + j]  = (_Float16)bp[(long)(16 + j) * HC];
        bf[12 + j] = (_Float16)bp[(long)(20 + j) * HC];
      }
      acc[t] = __builtin_amdgcn_wmma_f32_16x16x32_f16(
          false, af, false, bf, (short)0, acc[t], false, false);
    }
  }
  // ---- store: lane holds col n0+t*16+l16, rows r+8*hi (r=0..7)
#pragma unroll
  for (int t = 0; t < 4; ++t) {
    const int col = n0 + t * 16 + l16;
    float* dp = D + ((long)blockIdx.x * 16 + 8 * hi) * HC + col;
#pragma unroll
    for (int r = 0; r < 8; ++r) dp[(long)r * HC] = acc[t][r];
  }
}

// ---------------------------------------------------------------------------
// Fused edge GEMM: async-prefetch the 16x64 f32 A tile into LDS (ASYNCcnt),
// run WMMA from LDS fragments, stage the 16x256 result in LDS, then compute
// e_att[e,h] = dot(ef, att_edge[h]) and atomicAdd s_att[dst[e],h].
// ef is never written to global memory (saves ~1.6 GB of HBM traffic/layer).
// ---------------------------------------------------------------------------
__global__ __launch_bounds__(128)
void gemm_edge_fused_kernel(const float* __restrict__ A,      // [E,64]
                            const float* __restrict__ W,      // [64,256]
                            const float* __restrict__ att_e,  // [4,64]
                            const int*   __restrict__ dst,    // [E]
                            float* __restrict__ e_att,        // [E,4]
                            float* __restrict__ s_att) {      // [N,4] (atomic)
  __shared__ float atile[16 * EDIM];  // 4 KB  (async-prefetched A tile)
  __shared__ float tile[16 * HC];     // 16 KB (output staging)
  const int tid  = threadIdx.x;
  const int lane = tid & 31;
  const int wave = tid >> 5;
  const int hi   = lane >> 4;
  const int l16  = lane & 15;
  const int n0   = wave * 64;

  // ---- async bulk copy of the A tile: 1024 floats = 256 b128, 2 per thread
  {
    const float* g0 = A + (size_t)blockIdx.x * (16 * EDIM) + tid * 4;
    const float* g1 = g0 + 128 * 4;
    unsigned l0 = (unsigned)(unsigned long)&atile[tid * 4];
    unsigned l1 = (unsigned)(unsigned long)&atile[512 + tid * 4];
    asm volatile("global_load_async_to_lds_b128 %0, %1, off"
                 :: "v"(l0), "v"(g0) : "memory");
    asm volatile("global_load_async_to_lds_b128 %0, %1, off"
                 :: "v"(l1), "v"(g1) : "memory");
    asm volatile("s_wait_asynccnt 0x0" ::: "memory");
  }
  __syncthreads();

  v8f acc[4] = {};
#pragma unroll
  for (int kk = 0; kk < EDIM; kk += 32) {
    // ---- A fragment from LDS
    const float* ap = atile + l16 * EDIM + kk + hi * 8;
    f4 a0 = *(const f4*)(ap);
    f4 a1 = *(const f4*)(ap + 4);
    f4 a2 = *(const f4*)(ap + 16);
    f4 a3 = *(const f4*)(ap + 20);
    v16h af;
#pragma unroll
    for (int j = 0; j < 4; ++j) {
      af[j]      = (_Float16)a0[j];
      af[4 + j]  = (_Float16)a1[j];
      af[8 + j]  = (_Float16)a2[j];
      af[12 + j] = (_Float16)a3[j];
    }
    // ---- B fragments from global (L2-resident weights)
    const float* wp = W + (long)(kk + hi * 8) * HC + l16;
#pragma unroll
    for (int t = 0; t < 4; ++t) {
      const float* bp = wp + n0 + t * 16;
      v16h bf;
#pragma unroll
      for (int j = 0; j < 4; ++j) {
        bf[j]      = (_Float16)bp[(long)j * HC];
        bf[4 + j]  = (_Float16)bp[(long)(4 + j) * HC];
        bf[8 + j]  = (_Float16)bp[(long)(16 + j) * HC];
        bf[12 + j] = (_Float16)bp[(long)(20 + j) * HC];
      }
      acc[t] = __builtin_amdgcn_wmma_f32_16x16x32_f16(
          false, af, false, bf, (short)0, acc[t], false, false);
    }
  }
  // ---- stage 16x256 output tile into LDS
#pragma unroll
  for (int t = 0; t < 4; ++t) {
    const int col = n0 + t * 16 + l16;
#pragma unroll
    for (int r = 0; r < 8; ++r) tile[(8 * hi + r) * HC + col] = acc[t][r];
  }
  __syncthreads();
  // ---- epilogue: attention dot per (row, head) + atomic scatter of s_att
  if (tid < 64) {
    const int rrow = tid >> 2, hh = tid & 3;
    const long e = (long)blockIdx.x * 16 + rrow;
    const float* tr = tile + rrow * HC + hh * CC;
    const float* ae = att_e + hh * CC;
    float s = 0.f;
#pragma unroll 8
    for (int c = 0; c < CC; ++c) s += tr[c] * ae[c];
    e_att[e * NH + hh] = s;
    const int d = dst[e];
    atomicAdd(&s_att[(long)d * NH + hh], s);
  }
}

// ---------------------------------------------------------------------------
__global__ void count_kernel(const int* __restrict__ dst, float* __restrict__ cnt) {
  const long e = (long)blockIdx.x * blockDim.x + threadIdx.x;
  if (e < EE) atomicAdd(&cnt[dst[e]], 1.0f);
}

__global__ void node_att_kernel(const float* __restrict__ h,
                                const float* __restrict__ as_, const float* __restrict__ ad_,
                                float* __restrict__ a_src, float* __restrict__ a_dst) {
  const long idx = (long)blockIdx.x * blockDim.x + threadIdx.x;
  if (idx >= (long)NN * NH) return;
  const long n = idx >> 2;
  const int hh = idx & 3;
  const float* hp = h + n * HC + hh * CC;
  float ss = 0.f, sd = 0.f;
#pragma unroll 8
  for (int c = 0; c < CC; ++c) {
    const float v = hp[c];
    ss += v * as_[hh * CC + c];
    sd += v * ad_[hh * CC + c];
  }
  a_src[idx] = ss;
  a_dst[idx] = sd;
}

// self-loop logits; also initializes the segment max (every node has one)
__global__ void self_logit_kernel(const float* __restrict__ a_src, const float* __restrict__ a_dst,
                                  const float* __restrict__ s_att, const float* __restrict__ cnt,
                                  float* __restrict__ s_logit, unsigned* __restrict__ mkey) {
  const long idx = (long)blockIdx.x * blockDim.x + threadIdx.x;
  if (idx >= (long)NN * NH) return;
  const long n = idx >> 2;
  float cv = cnt[n];
  cv = cv < 1.f ? 1.f : cv;
  const float sl = lrelu(a_src[idx] + a_dst[idx] + s_att[idx] / cv);
  s_logit[idx] = sl;
  mkey[idx] = enc_ord(sl);
}

__global__ void edge_logit_kernel(const int* __restrict__ src, const int* __restrict__ dst,
                                  const float* __restrict__ a_src, const float* __restrict__ a_dst,
                                  const float* __restrict__ e_att,
                                  float* __restrict__ e_logit, unsigned* __restrict__ mkey) {
  const long idx = (long)blockIdx.x * blockDim.x + threadIdx.x;
  if (idx >= (long)EE * NH) return;
  const long e = idx >> 2;
  const int hh = idx & 3;
  const int s = src[e], d = dst[e];
  const float lg = lrelu(a_src[(long)s * NH + hh] + a_dst[(long)d * NH + hh] + e_att[idx]);
  e_logit[idx] = lg;
  atomicMax(&mkey[(long)d * NH + hh], enc_ord(lg));
}

__global__ void edge_exp_kernel(const int* __restrict__ dst,
                                float* __restrict__ e_logit,
                                const unsigned* __restrict__ mkey,
                                float* __restrict__ denom) {
  const long idx = (long)blockIdx.x * blockDim.x + threadIdx.x;
  if (idx >= (long)EE * NH) return;
  const long e = idx >> 2;
  const int hh = idx & 3;
  const int d = dst[e];
  const float p = __expf(e_logit[idx] - dec_ord(mkey[(long)d * NH + hh]));
  e_logit[idx] = p;   // overwrite with exp
  atomicAdd(&denom[(long)d * NH + hh], p);
}

__global__ void self_exp_kernel(float* __restrict__ s_logit,
                                const unsigned* __restrict__ mkey,
                                float* __restrict__ denom) {
  const long idx = (long)blockIdx.x * blockDim.x + threadIdx.x;
  if (idx >= (long)NN * NH) return;
  const float p = __expf(s_logit[idx] - dec_ord(mkey[idx]));
  s_logit[idx] = p;
  denom[idx] += p;    // non-atomic: runs after edge_exp_kernel (stream order)
}

// one wave per edge; lane covers 8 contiguous channels (one head per lane)
__global__ __launch_bounds__(256)
void scatter_edges_kernel(const int* __restrict__ src, const int* __restrict__ dst,
                          const float* __restrict__ p, const float* __restrict__ denom,
                          const float* __restrict__ h, float* __restrict__ out) {
  const int lane = threadIdx.x & 31;
  const long e = (long)blockIdx.x * 8 + (threadIdx.x >> 5);
  const int s = src[e], d = dst[e];
  const int head = lane >> 3;
  const float alpha = p[e * NH + head] / (denom[(long)d * NH + head] + 1e-16f);
  const float* hp = h + (long)s * HC + lane * 8;
  float* op = out + (long)d * HC + lane * 8;
  f4 h0 = *(const f4*)hp;
  f4 h1 = *(const f4*)(hp + 4);
#pragma unroll
  for (int j = 0; j < 4; ++j) {
    atomicAdd(op + j,     alpha * h0[j]);
    atomicAdd(op + 4 + j, alpha * h1[j]);
  }
}

__global__ void scatter_self_kernel(const float* __restrict__ ps, const float* __restrict__ denom,
                                    const float* __restrict__ h, float* __restrict__ out) {
  const long idx = (long)blockIdx.x * blockDim.x + threadIdx.x;
  if (idx >= (long)NN * HC) return;
  const long n = idx >> 8;
  const int col = idx & 255;
  const int head = col >> 6;
  const float alpha = ps[n * NH + head] / (denom[n * NH + head] + 1e-16f);
  out[idx] += alpha * h[idx];
}

__global__ void finalize1_kernel(const float* __restrict__ out1, const float* __restrict__ bias1,
                                 float* __restrict__ x2) {
  const long idx = (long)blockIdx.x * blockDim.x + threadIdx.x;
  if (idx >= (long)NN * HC) return;
  const int col = idx & 255;
  const float v = out1[idx] + bias1[col];
  x2[idx] = v > 0.f ? v : (__expf(v) - 1.f);   // ELU
}

__global__ void final2_kernel(const float* __restrict__ out2, const float* __restrict__ bias2,
                              float* __restrict__ y) {
  const long idx = (long)blockIdx.x * blockDim.x + threadIdx.x;
  if (idx >= (long)NN * CC) return;
  const long n = idx >> 6;
  const int c = idx & 63;
  const float* r = out2 + n * HC + c;
  y[idx] = 0.25f * (r[0] + r[64] + r[128] + r[192]) + bias2[c];
}

// ---------------------------------------------------------------------------
extern "C" void kernel_launch(void* const* d_in, const int* in_sizes, int n_in,
                              void* d_out, int out_size, void* d_ws, size_t ws_size,
                              hipStream_t stream) {
  (void)in_sizes; (void)n_in; (void)out_size; (void)ws_size;
  const float* x    = (const float*)d_in[0];   // [N,128]
  const float* efeat= (const float*)d_in[1];   // [E,64]
  const int*   eidx = (const int*)  d_in[2];   // [2,E]
  const int* src = eidx;
  const int* dst = eidx + EE;
  const float* W1  = (const float*)d_in[3];
  const float* as1 = (const float*)d_in[4];
  const float* ad1 = (const float*)d_in[5];
  const float* ae1 = (const float*)d_in[6];
  const float* b1  = (const float*)d_in[7];
  const float* eW1 = (const float*)d_in[8];
  const float* W2  = (const float*)d_in[9];
  const float* as2 = (const float*)d_in[10];
  const float* ad2 = (const float*)d_in[11];
  const float* ae2 = (const float*)d_in[12];
  const float* b2  = (const float*)d_in[13];
  const float* eW2 = (const float*)d_in[14];
  float* y = (float*)d_out;                    // [N,64]

  // workspace carve-up (256B aligned)
  size_t off = 0;
  auto carve = [&](size_t bytes) {
    void* p = (char*)d_ws + off;
    off += (bytes + 255) & ~(size_t)255;
    return p;
  };
  float*    B0     = (float*)   carve((size_t)NN * HC * 4);  // h1 -> x2 -> outacc2
  float*    B2     = (float*)   carve((size_t)NN * HC * 4);  // outacc1 -> h2
  float*    e_att  = (float*)   carve((size_t)EE * NH * 4);
  float*    e_lg   = (float*)   carve((size_t)EE * NH * 4);  // logits -> exp
  float*    a_src  = (float*)   carve((size_t)NN * NH * 4);
  float*    a_dst  = (float*)   carve((size_t)NN * NH * 4);
  float*    s_att  = (float*)   carve((size_t)NN * NH * 4);
  float*    s_lg   = (float*)   carve((size_t)NN * NH * 4);
  unsigned* mkey   = (unsigned*)carve((size_t)NN * NH * 4);
  float*    denom  = (float*)   carve((size_t)NN * NH * 4);
  float*    cnt    = (float*)   carve((size_t)NN * 4);

  const int T = 256;
  const int gN4  = (NN * NH + T - 1) / T;     // node x head
  const int gE4  = (EE * NH) / T;             // edge x head (exact)
  const int gNHC = (NN * HC) / T;             // node x 256 (exact)
  const int gNC  = (NN * CC) / T;             // node x 64 (exact)
  const int gE   = (EE + T - 1) / T;

  // ======================= layer 1 =======================
  hipMemsetAsync(cnt,   0, (size_t)NN * 4,        stream);
  hipMemsetAsync(s_att, 0, (size_t)NN * NH * 4,   stream);
  hipMemsetAsync(denom, 0, (size_t)NN * NH * 4,   stream);
  hipMemsetAsync(B2,    0, (size_t)NN * HC * 4,   stream);

  count_kernel<<<gE, T, 0, stream>>>(dst, cnt);
  gemm_node_kernel<<<NN / 16, 128, 0, stream>>>(x, W1, B0, FIN);          // h1 = x@W1
  gemm_edge_fused_kernel<<<EE / 16, 128, 0, stream>>>(efeat, eW1, ae1, dst, e_att, s_att);
  node_att_kernel<<<gN4, T, 0, stream>>>(B0, as1, ad1, a_src, a_dst);
  self_logit_kernel<<<gN4, T, 0, stream>>>(a_src, a_dst, s_att, cnt, s_lg, mkey);
  edge_logit_kernel<<<gE4, T, 0, stream>>>(src, dst, a_src, a_dst, e_att, e_lg, mkey);
  edge_exp_kernel<<<gE4, T, 0, stream>>>(dst, e_lg, mkey, denom);
  self_exp_kernel<<<gN4, T, 0, stream>>>(s_lg, mkey, denom);
  scatter_edges_kernel<<<EE / 8, 256, 0, stream>>>(src, dst, e_lg, denom, B0, B2);
  scatter_self_kernel<<<gNHC, T, 0, stream>>>(s_lg, denom, B0, B2);
  finalize1_kernel<<<gNHC, T, 0, stream>>>(B2, b1, B0);                   // B0 := elu(out1+b1)

  // ======================= layer 2 =======================
  hipMemsetAsync(s_att, 0, (size_t)NN * NH * 4, stream);
  hipMemsetAsync(denom, 0, (size_t)NN * NH * 4, stream);

  gemm_node_kernel<<<NN / 16, 128, 0, stream>>>(B0, W2, B2, HC);          // h2 = x2@W2
  gemm_edge_fused_kernel<<<EE / 16, 128, 0, stream>>>(efeat, eW2, ae2, dst, e_att, s_att);
  hipMemsetAsync(B0, 0, (size_t)NN * HC * 4, stream);                     // outacc2 (after x2 read)
  node_att_kernel<<<gN4, T, 0, stream>>>(B2, as2, ad2, a_src, a_dst);
  self_logit_kernel<<<gN4, T, 0, stream>>>(a_src, a_dst, s_att, cnt, s_lg, mkey);
  edge_logit_kernel<<<gE4, T, 0, stream>>>(src, dst, a_src, a_dst, e_att, e_lg, mkey);
  edge_exp_kernel<<<gE4, T, 0, stream>>>(dst, e_lg, mkey, denom);
  self_exp_kernel<<<gN4, T, 0, stream>>>(s_lg, mkey, denom);
  scatter_edges_kernel<<<EE / 8, 256, 0, stream>>>(src, dst, e_lg, denom, B2, B0);
  scatter_self_kernel<<<gNHC, T, 0, stream>>>(s_lg, denom, B2, B0);
  final2_kernel<<<gNC, T, 0, stream>>>(B0, b2, y);                        // mean heads + bias
}
